// EnhancedAttentionBlock_52587579572726
// MI455X (gfx1250) — matrix operand order repaired and checked by reading, not compiled, not run on hardware
//
#include <hip/hip_runtime.h>

// ---------------------------------------------------------------------------
// EnhancedAttentionBlock on MI455X (gfx1250): bf16 WMMA everywhere, f32
// accumulate/softmax/norms.  B=2, C=256, H=W=32, N=1024 tokens, 16 heads d=16.
// Rotation branch reduced to a single pass (attention is permutation-
// equivariant under rot90).  Weights are pre-swizzled into the WMMA B-operand
// lane layout so the GEMM inner loop is pure vector loads + v_wmma.
// ---------------------------------------------------------------------------

typedef __bf16 bf16_t;
typedef bf16_t   v16bf __attribute__((ext_vector_type(16)));
typedef float    v8f   __attribute__((ext_vector_type(8)));
typedef unsigned v8u   __attribute__((ext_vector_type(8)));

__device__ __forceinline__ bf16_t f2bf(float f) {
  unsigned u = __builtin_bit_cast(unsigned, f);
  u += 0x7FFFu + ((u >> 16) & 1u);                  // round-to-nearest-even
  unsigned short h = (unsigned short)(u >> 16);
  return __builtin_bit_cast(bf16_t, h);
}
__device__ __forceinline__ v8f wmma_bf(v16bf a, v16bf b, v8f c) {
  return __builtin_amdgcn_wmma_f32_16x16x32_bf16(false, a, false, b, (short)0, c,
                                                 false, false);
}

// ---- WMMA operand builders (CDNA5 16x16x32 bf16 layouts, wave32) ----------
// A 16x32: lane<16 -> M=lane, slots0..7 K=0..7, slots8..15 K=16..23
//          lane>=16 -> M=lane-16, slots0..7 K=8..15, slots8..15 K=24..31
// K-pairs within a VGPR are contiguous -> build from 8 dword loads.
__device__ __forceinline__ v16bf gather_A32(const bf16_t* A, int lda, int m0, int kc) {
  int lane = threadIdx.x & 31, half = lane >> 4, m = m0 + (lane & 15);
  const unsigned* row32 = (const unsigned*)(A + (size_t)m * lda + kc);
  v8u t;
#pragma unroll
  for (int j = 0; j < 8; ++j) {
    int kp = ((j < 4) ? j : (j + 4)) + half * 4;   // dword index = pair_base/2
    t[j] = row32[kp];
  }
  return __builtin_bit_cast(v16bf, t);
}
// B 32x16 pre-swizzled: fragment for (ktile,ntile) is 32 lanes x 16 bf16,
// lane-contiguous -> one aligned 32-byte load.
__device__ __forceinline__ v16bf load_Bsw(const bf16_t* Bsw, int Ntiles, int kt, int nt) {
  int lane = threadIdx.x & 31;
  return *(const v16bf*)(Bsw + (((size_t)kt * Ntiles + nt) * 32 + lane) * 16);
}
// C/D 16x16 f32: lane<16 -> N=lane, vgpr r -> M=r ; lane>=16 -> M=r+8

// ---------------------------------------------------------------------------
// f32 -> bf16 array convert
__global__ void f2bf_k(const float* __restrict__ s, bf16_t* __restrict__ d, int n) {
  int i = blockIdx.x * 256 + threadIdx.x;
  if (i < n) d[i] = f2bf(s[i]);
}

// convert + swizzle a (K x N) f32 weight into WMMA B-operand fragments:
// out[((ktile*Ntiles + ntile)*32 + lane)*16 + slot] = W[k, n]
//   k = ktile*32 + (slot&7) + (lane>>4)*8 + (slot>>3)*16,  n = ntile*16 + (lane&15)
__global__ void swz_b_k(const float* __restrict__ W, bf16_t* __restrict__ out,
                        int K, int N) {
  int idx = blockIdx.x * 256 + threadIdx.x;
  if (idx >= K * N) return;
  int e = idx & 15;
  int l = (idx >> 4) & 31;
  int blk = idx >> 9;
  int Ntiles = N >> 4;
  int ntile = blk % Ntiles, ktile = blk / Ntiles;
  int k = ktile * 32 + (e & 7) + (l >> 4) * 8 + (e >> 3) * 16;
  int n = ntile * 16 + (l & 15);
  out[idx] = f2bf(W[(size_t)k * N + n]);
}

// group stats of x in (B,C,N) layout: group = 8 contiguous channels -> 8192
// contiguous floats.  st[2*bg] = mean, st[2*bg+1] = rsqrt(var+eps)
__global__ void stats_x_k(const float* __restrict__ x, float* __restrict__ st) {
  int bg = blockIdx.x;
  const float* base = x + (size_t)bg * 8192;
  float s = 0.f, ss = 0.f;
  for (int i = threadIdx.x; i < 8192; i += 256) {
    float v = base[i];
    s += v; ss += v * v;
  }
  __shared__ float rs[256], rq[256];
  rs[threadIdx.x] = s; rq[threadIdx.x] = ss;
  __syncthreads();
  for (int o = 128; o > 0; o >>= 1) {
    if (threadIdx.x < o) { rs[threadIdx.x] += rs[threadIdx.x + o]; rq[threadIdx.x] += rq[threadIdx.x + o]; }
    __syncthreads();
  }
  if (threadIdx.x == 0) {
    float m = rs[0] * (1.f / 8192.f);
    float var = rq[0] * (1.f / 8192.f) - m * m;
    st[2 * bg] = m;
    st[2 * bg + 1] = rsqrtf(var + 1e-5f);
  }
}

// group stats of y in token-major (2048 x 256)
__global__ void stats_y_k(const float* __restrict__ y, float* __restrict__ st) {
  int bg = blockIdx.x, b = bg >> 5, g = bg & 31;
  float s = 0.f, ss = 0.f;
  for (int i = threadIdx.x; i < 8192; i += 256) {
    int p = i >> 3, cc = i & 7;
    float v = y[((size_t)b * 1024 + p) * 256 + g * 8 + cc];
    s += v; ss += v * v;
  }
  __shared__ float rs[256], rq[256];
  rs[threadIdx.x] = s; rq[threadIdx.x] = ss;
  __syncthreads();
  for (int o = 128; o > 0; o >>= 1) {
    if (threadIdx.x < o) { rs[threadIdx.x] += rs[threadIdx.x + o]; rq[threadIdx.x] += rq[threadIdx.x + o]; }
    __syncthreads();
  }
  if (threadIdx.x == 0) {
    float m = rs[0] * (1.f / 8192.f);
    float var = rq[0] * (1.f / 8192.f) - m * m;
    st[2 * bg] = m;
    st[2 * bg + 1] = rsqrtf(var + 1e-5f);
  }
}

// token-major bf16: raw x, ra-gnorm(x), g-gnorm(x)
__global__ void norm_x_k(const float* __restrict__ x, const float* __restrict__ st,
                         const float* __restrict__ raw, const float* __restrict__ rab,
                         const float* __restrict__ gnw, const float* __restrict__ gnb,
                         bf16_t* __restrict__ xT, bf16_t* __restrict__ xnra,
                         bf16_t* __restrict__ xng) {
  int t = blockIdx.x, c = threadIdx.x;
  int b = t >> 10, p = t & 1023;
  float xv = x[(size_t)b * 262144 + (size_t)c * 1024 + p];
  int g = c >> 3;
  float m = st[2 * (b * 32 + g)], rs = st[2 * (b * 32 + g) + 1];
  float xn = (xv - m) * rs;
  size_t o = (size_t)t * 256 + c;
  xT[o]   = f2bf(xv);
  xnra[o] = f2bf(xn * raw[c] + rab[c]);
  xng[o]  = f2bf(xn * gnw[c] + gnb[c]);
}

// halo patches: 32 blocks x 224 tokens (196 real + 28 zero pad) x 256 ch, bf16
__global__ void halo_gather_k(const float* __restrict__ x, bf16_t* __restrict__ P) {
  int row = blockIdx.x, c = threadIdx.x;
  int blk = row / 224, tok = row - blk * 224;
  float v = 0.f;
  if (tok < 196) {
    int r = tok / 14, cn = tok - r * 14;
    int b = blk >> 4, hb = (blk >> 2) & 3, wb = blk & 3;
    int py = hb * 8 - 3 + r, px = wb * 8 - 3 + cn;
    if (py >= 0 && py < 32 && px >= 0 && px < 32)
      v = x[(size_t)b * 262144 + (size_t)c * 1024 + py * 32 + px];
  }
  P[(size_t)row * 256 + c] = f2bf(v);
}

// ---------------------------------------------------------------------------
// generic WMMA GEMM -> bf16 out, optional bias + relu.  grid(Nt, Mt), 32 thr.
__global__ void __launch_bounds__(32)
gemm_bf16out_k(const bf16_t* __restrict__ A, int lda, const bf16_t* __restrict__ Bsw,
               const float* __restrict__ bias, bf16_t* __restrict__ outp,
               int ldo, int K, int act) {
  int nt = blockIdx.x, mt = blockIdx.y, Ntiles = gridDim.x;
  int lane = threadIdx.x & 31, half = lane >> 4, ln = lane & 15;
  v8f acc = {};
  for (int kt = 0; kt < (K >> 5); ++kt)
    acc = wmma_bf(gather_A32(A, lda, mt * 16, kt * 32), load_Bsw(Bsw, Ntiles, kt, nt), acc);
  int n = nt * 16 + ln;
  float bv = bias ? bias[n] : 0.f;
#pragma unroll
  for (int r = 0; r < 8; ++r) {
    int m = mt * 16 + r + half * 8;
    float v = acc[r] + bv;
    if (act == 1) v = fmaxf(v, 0.f);
    outp[(size_t)m * ldo + n] = f2bf(v);
  }
}

// WMMA GEMM with f32 epilogues.
// mode 1: ro     : out[token*256+n] = v + bias + x[b,n,p]
// mode 2: halo   : remap rows (blk,center j)->token; out[tok*256+n] = v + bias
// mode 3: final  : out[b,n,p] = v + bias + x[b,n,p]   (NCHW)
// mode 4: spatial: out[token*256+n] = x[b,n,p] * (1 + sigmoid(v + bias))
__global__ void __launch_bounds__(32)
gemm_f32ep_k(const bf16_t* __restrict__ A, int lda, const bf16_t* __restrict__ Bsw,
             const float* __restrict__ bias, const float* __restrict__ xin,
             float* __restrict__ outp, int K, int mode) {
  int nt = blockIdx.x, mt = blockIdx.y, Ntiles = gridDim.x;
  int lane = threadIdx.x & 31, half = lane >> 4, ln = lane & 15;
  v8f acc = {};
  for (int kt = 0; kt < (K >> 5); ++kt)
    acc = wmma_bf(gather_A32(A, lda, mt * 16, kt * 32), load_Bsw(Bsw, Ntiles, kt, nt), acc);
  int n = nt * 16 + ln;
  float bv = bias ? bias[n] : 0.f;
#pragma unroll
  for (int r = 0; r < 8; ++r) {
    int m = mt * 16 + r + half * 8;
    float v = acc[r] + bv;
    if (mode == 1) {
      int b = m >> 10, p = m & 1023;
      outp[(size_t)m * 256 + n] = v + xin[(size_t)b * 262144 + (size_t)n * 1024 + p];
    } else if (mode == 2) {
      int blk = m >> 6, j = m & 63;
      int b = blk >> 4, hb = (blk >> 2) & 3, wb = blk & 3;
      int rr = j >> 3, cc = j & 7;
      int p = (hb * 8 + rr) * 32 + wb * 8 + cc;
      outp[((size_t)b * 1024 + p) * 256 + n] = v;
    } else if (mode == 3) {
      int b = m >> 10, p = m & 1023;
      size_t xi = (size_t)b * 262144 + (size_t)n * 1024 + p;
      outp[xi] = v + xin[xi];
    } else {
      int b = m >> 10, p = m & 1023;
      float xv = xin[(size_t)b * 262144 + (size_t)n * 1024 + p];
      float sg = 1.f / (1.f + __expf(-v));
      outp[(size_t)m * 256 + n] = xv * (1.f + sg);
    }
  }
}

// ---------------------------------------------------------------------------
// Flash attention core (one wave, one 16-row q-tile).  qkv rows hold
// [q(256) | k(256) | v(256)], d=16 padded to K=32 for score WMMAs.
__device__ __forceinline__ v8f flash_attn_core(const bf16_t* __restrict__ qkv,
                                               long kvBase, long qrow, int head,
                                               int KVP, int KVvalid) {
  __shared__ bf16_t pbuf[16 * 32];
  const float scale = 0.25f;  // d^-0.5, d=16
  int lane = threadIdx.x & 31, half = lane >> 4, ln = lane & 15;
  const int hq = head * 16;
  // q fragment: lanes<16 slots 0..15 = K 0..15 (dword pairs), lanes>=16 zero padding
  v16bf aq;
  {
    const unsigned* q32 = (const unsigned*)(qkv + qrow * 768 + hq);
    v8u t = {};
#pragma unroll
    for (int j = 0; j < 8; ++j) t[j] = half ? 0u : q32[j];
    aq = __builtin_bit_cast(v16bf, t);
  }
  v8f oacc = {};
  float mrow[8], lrow[8];
#pragma unroll
  for (int r = 0; r < 8; ++r) { mrow[r] = -1e30f; lrow[r] = 0.f; }

  for (int kc = 0; kc < KVP; kc += 32) {
    v8f s0 = {}, s1 = {};
    v8u zz = {};
    // k^T fragments: half<16 lanes hold the 16 contiguous features of their
    // kv token -> one aligned 32B vector load; upper half is the K-padding.
    v16bf bk0 = __builtin_bit_cast(v16bf, zz), bk1 = __builtin_bit_cast(v16bf, zz);
    if (half == 0) {
      bk0 = *(const v16bf*)(qkv + (kvBase + kc + ln) * 768 + 256 + hq);
      bk1 = *(const v16bf*)(qkv + (kvBase + kc + 16 + ln) * 768 + 256 + hq);
    }
    s0 = wmma_bf(aq, bk0, s0);
    s1 = wmma_bf(aq, bk1, s1);

    int c0 = kc + ln, c1 = kc + 16 + ln;
    float t0[8], t1[8], cm[8];
#pragma unroll
    for (int r = 0; r < 8; ++r) {
      t0[r] = (c0 < KVvalid) ? s0[r] * scale : -1e30f;
      t1[r] = (c1 < KVvalid) ? s1[r] * scale : -1e30f;
      cm[r] = fmaxf(t0[r], t1[r]);
    }
#pragma unroll
    for (int off = 1; off < 16; off <<= 1)
#pragma unroll
      for (int r = 0; r < 8; ++r) cm[r] = fmaxf(cm[r], __shfl_xor(cm[r], off, 32));
    float fac[8], cs[8], p0[8], p1[8];
#pragma unroll
    for (int r = 0; r < 8; ++r) {
      float mn = fmaxf(mrow[r], cm[r]);
      fac[r] = __expf(mrow[r] - mn);
      mrow[r] = mn;
      p0[r] = __expf(t0[r] - mn);
      p1[r] = __expf(t1[r] - mn);
      cs[r] = p0[r] + p1[r];
    }
#pragma unroll
    for (int off = 1; off < 16; off <<= 1)
#pragma unroll
      for (int r = 0; r < 8; ++r) cs[r] += __shfl_xor(cs[r], off, 32);
#pragma unroll
    for (int r = 0; r < 8; ++r) { lrow[r] = lrow[r] * fac[r] + cs[r]; oacc[r] *= fac[r]; }

    __syncthreads();
#pragma unroll
    for (int r = 0; r < 8; ++r) {
      pbuf[(r + half * 8) * 32 + ln] = f2bf(p0[r]);
      pbuf[(r + half * 8) * 32 + 16 + ln] = f2bf(p1[r]);
    }
    __syncthreads();
    // probs as A operand: dword-pair gathers from LDS
    v16bf ap;
    {
      const unsigned* p32 = (const unsigned*)pbuf;
      v8u t;
#pragma unroll
      for (int j = 0; j < 8; ++j) {
        int kp = ((j < 4) ? j : (j + 4)) + half * 4;
        t[j] = p32[ln * 16 + kp];
      }
      ap = __builtin_bit_cast(v16bf, t);
    }
    v16bf bv;
#pragma unroll
    for (int s = 0; s < 16; ++s)
      bv[s] = qkv[(kvBase + kc + (s & 7) + half * 8 + (s >> 3) * 16) * 768 + 512 + hq + ln];
    oacc = wmma_bf(ap, bv, oacc);
  }
  v8f o;
#pragma unroll
  for (int r = 0; r < 8; ++r) o[r] = oacc[r] / lrow[r];
  return o;
}

// flat attention over 1024 tokens (rotation & global paths). grid(64,16,B)
__global__ void __launch_bounds__(32)
attn_flat_k(const bf16_t* __restrict__ qkv, float* __restrict__ outp) {
  int qt = blockIdx.x, h = blockIdx.y, b = blockIdx.z;
  int lane = threadIdx.x & 31, half = lane >> 4, ln = lane & 15;
  long base = (long)b * 1024;
  v8f o = flash_attn_core(qkv, base, base + qt * 16 + ln, h, 1024, 1024);
#pragma unroll
  for (int r = 0; r < 8; ++r) {
    long m = base + qt * 16 + r + half * 8;
    outp[m * 256 + h * 16 + ln] = o[r];
  }
}

// halo attention: q = 64 center tokens, kv = 224 (196 valid). grid(4,16,32)
__global__ void __launch_bounds__(32)
attn_halo_k(const bf16_t* __restrict__ qkv, float* __restrict__ outp) {
  int qt = blockIdx.x, h = blockIdx.y, blk = blockIdx.z;
  int lane = threadIdx.x & 31, half = lane >> 4, ln = lane & 15;
  int j = qt * 16 + ln;
  int tok = (3 + (j >> 3)) * 14 + 3 + (j & 7);
  long base = (long)blk * 224;
  v8f o = flash_attn_core(qkv, base, base + tok, h, 224, 196);
#pragma unroll
  for (int r = 0; r < 8; ++r) {
    long m = (long)blk * 64 + qt * 16 + r + half * 8;
    outp[m * 256 + h * 16 + ln] = o[r];
  }
}

// combine: sp (gnorm of y) * w0 + ha*w1 + qo*w2 + ro*w3 -> bf16
__global__ void combine_k(const float* __restrict__ y, const float* __restrict__ sty,
                          const float* __restrict__ sgw, const float* __restrict__ sgb,
                          const float* __restrict__ ha, const float* __restrict__ qo,
                          const float* __restrict__ ro, const float* __restrict__ aw,
                          bf16_t* __restrict__ comb) {
  int t = blockIdx.x, c = threadIdx.x;
  int b = t >> 10, g = c >> 3;
  float m = sty[2 * (b * 32 + g)], rs = sty[2 * (b * 32 + g) + 1];
  size_t o = (size_t)t * 256 + c;
  float sp = (y[o] - m) * rs * sgw[c] + sgb[c];
  float a0 = aw[0], a1 = aw[1], a2 = aw[2], a3 = aw[3];
  float mx = fmaxf(fmaxf(a0, a1), fmaxf(a2, a3));
  float e0 = __expf(a0 - mx), e1 = __expf(a1 - mx), e2 = __expf(a2 - mx), e3 = __expf(a3 - mx);
  float inv = 1.f / (e0 + e1 + e2 + e3);
  comb[o] = f2bf((sp * e0 + ha[o] * e1 + qo[o] * e2 + ro[o] * e3) * inv);
}

// ---------------------------------------------------------------------------
extern "C" void kernel_launch(void* const* d_in, const int* in_sizes, int n_in,
                              void* d_out, int out_size, void* d_ws, size_t ws_size,
                              hipStream_t stream) {
  (void)in_sizes; (void)n_in; (void)out_size; (void)ws_size;
  const float* x        = (const float*)d_in[0];
  const float* sa_w1    = (const float*)d_in[1];
  const float* sa_b1    = (const float*)d_in[2];
  const float* sa_w2    = (const float*)d_in[3];
  const float* sa_b2    = (const float*)d_in[4];
  const float* sa_gn_w  = (const float*)d_in[5];
  const float* sa_gn_b  = (const float*)d_in[6];
  const float* ha_qkv_w = (const float*)d_in[7];
  const float* ha_pw    = (const float*)d_in[8];
  const float* ha_pb    = (const float*)d_in[9];
  const float* ra_gn_w  = (const float*)d_in[10];
  const float* ra_gn_b  = (const float*)d_in[11];
  const float* ra_qkv_w = (const float*)d_in[12];
  const float* ra_pw    = (const float*)d_in[13];
  const float* ra_pb    = (const float*)d_in[14];
  const float* gn_w     = (const float*)d_in[15];
  const float* gn_b     = (const float*)d_in[16];
  const float* toqkv_w  = (const float*)d_in[17];
  const float* toqkv_b  = (const float*)d_in[18];
  const float* attn_w   = (const float*)d_in[19];
  const float* proj_w   = (const float*)d_in[20];
  const float* proj_b   = (const float*)d_in[21];
  float* out = (float*)d_out;

  char* ws = (char*)d_ws;
  size_t off = 0;
  auto take = [&](size_t bytes) -> void* {
    void* p = ws + off;
    off = (off + bytes + 255) & ~(size_t)255;
    return p;
  };
  bf16_t* w_saw1   = (bf16_t*)take(256 * 32 * 2);
  bf16_t* w_saw2   = (bf16_t*)take(32 * 256 * 2);
  bf16_t* w_haqkv  = (bf16_t*)take(256 * 768 * 2);
  bf16_t* w_raqkv  = (bf16_t*)take(256 * 768 * 2);
  bf16_t* w_toqkv  = (bf16_t*)take(256 * 768 * 2);
  bf16_t* w_haproj = (bf16_t*)take(256 * 256 * 2);
  bf16_t* w_raproj = (bf16_t*)take(256 * 256 * 2);
  bf16_t* w_proj   = (bf16_t*)take(256 * 256 * 2);
  bf16_t* xT      = (bf16_t*)take((size_t)2048 * 256 * 2);
  bf16_t* xnra    = (bf16_t*)take((size_t)2048 * 256 * 2);
  bf16_t* xng     = (bf16_t*)take((size_t)2048 * 256 * 2);
  bf16_t* s1b     = (bf16_t*)take((size_t)2048 * 32 * 2);
  bf16_t* qkv_ra  = (bf16_t*)take((size_t)2048 * 768 * 2);
  bf16_t* qkv_g   = (bf16_t*)take((size_t)2048 * 768 * 2);
  bf16_t* Pb      = (bf16_t*)take((size_t)7168 * 256 * 2);
  bf16_t* qkv_h   = (bf16_t*)take((size_t)7168 * 768 * 2);
  bf16_t* o_ra_bf = (bf16_t*)take((size_t)2048 * 256 * 2);
  bf16_t* o_h_bf  = (bf16_t*)take((size_t)2048 * 256 * 2);
  bf16_t* combb   = (bf16_t*)take((size_t)2048 * 256 * 2);
  float* stats_x = (float*)take(128 * 4);
  float* stats_y = (float*)take(128 * 4);
  float* o_ra = (float*)take((size_t)2048 * 256 * 4);
  float* o_g  = (float*)take((size_t)2048 * 256 * 4);
  float* o_h  = (float*)take((size_t)2048 * 256 * 4);
  float* yb   = (float*)take((size_t)2048 * 256 * 4);
  float* ro_t = (float*)take((size_t)2048 * 256 * 4);
  float* ha_t = (float*)take((size_t)2048 * 256 * 4);

  auto cvt = [&](const float* s, bf16_t* d, int n) {
    f2bf_k<<<(n + 255) / 256, 256, 0, stream>>>(s, d, n);
  };
  auto swz = [&](const float* s, bf16_t* d, int K, int N) {
    swz_b_k<<<(K * N + 255) / 256, 256, 0, stream>>>(s, d, K, N);
  };
  // weights -> bf16, pre-swizzled into WMMA B fragments
  swz(sa_w1, w_saw1, 256, 32);
  swz(sa_w2, w_saw2, 32, 256);
  swz(ha_qkv_w, w_haqkv, 256, 768);
  swz(ra_qkv_w, w_raqkv, 256, 768);
  swz(toqkv_w, w_toqkv, 256, 768);
  swz(ha_pw, w_haproj, 256, 256);
  swz(ra_pw, w_raproj, 256, 256);
  swz(proj_w, w_proj, 256, 256);

  // shared group stats of x; normalized token-major tensors
  stats_x_k<<<64, 256, 0, stream>>>(x, stats_x);
  norm_x_k<<<2048, 256, 0, stream>>>(x, stats_x, ra_gn_w, ra_gn_b, gn_w, gn_b,
                                     xT, xnra, xng);
  // halo patch gather (zero-padded to 224 tokens per block)
  halo_gather_k<<<7168, 256, 0, stream>>>(x, Pb);

  // QKV GEMMs
  gemm_bf16out_k<<<dim3(48, 128), 32, 0, stream>>>(xnra, 256, w_raqkv,
                                                   nullptr, qkv_ra, 768, 256, 0);
  gemm_bf16out_k<<<dim3(48, 128), 32, 0, stream>>>(xng, 256, w_toqkv,
                                                   toqkv_b, qkv_g, 768, 256, 0);
  gemm_bf16out_k<<<dim3(48, 448), 32, 0, stream>>>(Pb, 256, w_haqkv,
                                                   nullptr, qkv_h, 768, 256, 0);

  // attention (flash, WMMA)
  attn_flat_k<<<dim3(64, 16, 2), 32, 0, stream>>>(qkv_ra, o_ra);
  attn_flat_k<<<dim3(64, 16, 2), 32, 0, stream>>>(qkv_g, o_g);
  attn_halo_k<<<dim3(4, 16, 32), 32, 0, stream>>>(qkv_h, o_h);

  cvt(o_ra, o_ra_bf, 2048 * 256);
  cvt(o_h, o_h_bf, 2048 * 256);

  // rotation proj (+bias +x residual) and halo proj (+bias, scatter to tokens)
  gemm_f32ep_k<<<dim3(16, 128), 32, 0, stream>>>(o_ra_bf, 256, w_raproj,
                                                 ra_pb, x, ro_t, 256, 1);
  gemm_f32ep_k<<<dim3(16, 128), 32, 0, stream>>>(o_h_bf, 256, w_haproj,
                                                 ha_pb, x, ha_t, 256, 2);

  // spatial gate: relu(x@w1+b1) -> sigmoid(.@w2+b2); y = x*(1+sig)
  gemm_bf16out_k<<<dim3(2, 128), 32, 0, stream>>>(xT, 256, w_saw1, sa_b1,
                                                  s1b, 32, 256, 1);
  gemm_f32ep_k<<<dim3(16, 128), 32, 0, stream>>>(s1b, 32, w_saw2, sa_b2,
                                                 x, yb, 32, 4);
  stats_y_k<<<64, 256, 0, stream>>>(yb, stats_y);

  // weighted combine -> bf16
  combine_k<<<2048, 256, 0, stream>>>(yb, stats_y, sa_gn_w, sa_gn_b, ha_t, o_g,
                                      ro_t, attn_w, combb);

  // final projection (+bias +x) straight into d_out (NCHW)
  gemm_f32ep_k<<<dim3(16, 128), 32, 0, stream>>>(combb, 256, w_proj,
                                                 proj_b, x, out, 256, 3);
}
